// Attention_GLM_Wrapped_32238024524365
// MI455X (gfx1250) — matrix operand order repaired and checked by reading, not compiled, not run on hardware
//
#include <hip/hip_runtime.h>

// ---------------------------------------------------------------------------
// Types matching gfx1250 WMMA / TDM builtin signatures
// ---------------------------------------------------------------------------
typedef __attribute__((ext_vector_type(16))) __bf16         v16bf;
typedef __attribute__((ext_vector_type(8)))  float          v8f;
typedef __attribute__((ext_vector_type(4)))  unsigned int   u32x4;
typedef __attribute__((ext_vector_type(4)))  unsigned short u16x4;
typedef __attribute__((ext_vector_type(4)))  float          f32x4;
typedef __attribute__((ext_vector_type(8)))  int            i32x8;
typedef __attribute__((ext_vector_type(4)))  int            i32x4;

#if defined(__has_builtin)
#  if __has_builtin(__builtin_amdgcn_tensor_load_to_lds)
#    define HAVE_TDM 1
#  endif
#endif
#ifndef HAVE_TDM
#  define HAVE_TDM 0
#endif

union BF16x16 { v16bf v; u32x4 q[2]; };

__device__ __forceinline__ unsigned short f2bf(float f) {
  unsigned u = __builtin_bit_cast(unsigned, f);
  u += 0x7FFFu + ((u >> 16) & 1u);          // round-to-nearest-even
  return (unsigned short)(u >> 16);
}
__device__ __forceinline__ float bf2f(unsigned short h) {
  unsigned u = ((unsigned)h) << 16;
  return __builtin_bit_cast(float, u);
}

__device__ __forceinline__ v8f wmma_bf16(v16bf a, v16bf b, v8f c) {
  return __builtin_amdgcn_wmma_f32_16x16x32_bf16(
      false, a, false, b, (short)0, c, false, false);
}

// A-fragment: row owned by lane%16; K elems [kk + g*8, +8) and [kk+16+g*8, +8)
__device__ __forceinline__ v16bf load_afrag(const unsigned short* rowp, int kk, int g) {
  BF16x16 t;
  t.q[0] = *(const u32x4*)(rowp + kk + g * 8);
  t.q[1] = *(const u32x4*)(rowp + kk + 16 + g * 8);
  return t.v;
}
// B-fragment: column owned by lane%16; K elems [kk + g*8, +8) and [kk+16+g*8, +8)
__device__ __forceinline__ v16bf load_bfrag(const unsigned short* rowp, int kk, int g) {
  const unsigned short* s = rowp + kk + g * 8;
  BF16x16 t;
  t.q[0] = *(const u32x4*)(s);
  t.q[1] = *(const u32x4*)(s + 16);
  return t.v;
}

// ---------------------------------------------------------------------------
// TDM: async DMA of a 2D bf16 tile (rows x tile_k, row stride stride_elems)
// into LDS with per-row padding. D# per CDNA5 ISA.
// pad_interval code c -> pad every (8<<c) bytes; pad_amount code c -> c+1 DWORDs
// ---------------------------------------------------------------------------
#if HAVE_TDM
__device__ __forceinline__ void tdm_load_2d_bf16(
    unsigned lds_byte_addr, const void* gptr, int tile_k, int rows,
    int stride_elems, int pad_interval, int pad_amount) {
  unsigned long long ga = (unsigned long long)gptr;
  u32x4 g0;
  g0.x = 0x1u;                                   // count=1, user descriptor
  g0.y = lds_byte_addr;                          // lds_addr
  g0.z = (unsigned)ga;                           // global_addr[31:0]
  g0.w = (unsigned)((ga >> 32) & 0x01FFFFFFu)    // global_addr[56:32]
       | 0x80000000u;                            // type=2 ("image")
  i32x8 g1;
  g1[0] = (1 << 16)                 // data_size = 2 bytes
        | (1 << 20)                 // pad_enable
        | (pad_interval << 22)
        | (pad_amount << 25);
  g1[1] = (tile_k & 0xFFFF) << 16;  // atomic_addr=0 | tensor_dim0.lo16
  g1[2] = (rows & 0xFFFF) << 16;    // tensor_dim0.hi=0 | tensor_dim1.lo16
  g1[3] = (tile_k & 0xFFFF) << 16;  // tensor_dim1.hi=0 | tile_dim0
  g1[4] = rows & 0xFFFF;            // tile_dim1 | tile_dim2=0
  g1[5] = stride_elems;             // tensor_dim0_stride (lo32)
  g1[6] = 0;
  g1[7] = 0;
  i32x4 z4{};
  i32x8 z8{};
  __builtin_amdgcn_tensor_load_to_lds(g0, g1, z4, z4, z8, 0);
}
#endif

// ---------------------------------------------------------------------------
// Problem constants
// ---------------------------------------------------------------------------
#define SEQ    2048
#define BATCH  2
#define DMODEL 4096
#define NHEADS 32
#define HDIM   128
#define MROWS  (SEQ * BATCH)     // 4096
#define QKVN   (3 * DMODEL)      // 12288

// ---------------------------------------------------------------------------
// Kernel 0: fp32 -> bf16 bulk conversion (one-time; makes weights L2-resident
// at half size and lets TDM feed the GEMMs directly)
// ---------------------------------------------------------------------------
__global__ __launch_bounds__(256) void cvt_f32_bf16(
    const float* __restrict__ in, unsigned short* __restrict__ out) {
  size_t i = ((size_t)blockIdx.x * 256 + threadIdx.x) * 4;
  f32x4 fv = *(const f32x4*)&in[i];
  u16x4 hv;
  hv.x = f2bf(fv.x); hv.y = f2bf(fv.y); hv.z = f2bf(fv.z); hv.w = f2bf(fv.w);
  *(u16x4*)&out[i] = hv;
}

// ---------------------------------------------------------------------------
// Kernel 1/4: tiled bf16 WMMA GEMM  C[M,N] = A[M,K] * W[N,K]^T + bias
// A, W both bf16. Tiles staged by TDM, double-buffered in LDS.
// ---------------------------------------------------------------------------
#define KC 64
#define LDK 80            // LDS row stride in halves (160B, 16B aligned)
#define GTB (128 * LDK)   // one GEMM tile in halves (20480 B)

template <bool OUT_BF16>
__global__ __launch_bounds__(256) void gemm_wmma(
    const unsigned short* __restrict__ A, const unsigned short* __restrict__ Wb,
    const float* __restrict__ bias, void* __restrict__ Cptr,
    int M, int N, int K) {
  extern __shared__ __align__(16) unsigned short smem[];
  // layout: [A0][B0][A1][B1], each GTB halves

  const int tid  = threadIdx.x;
  const int wave = tid >> 5;
  const int lane = tid & 31;
  const int g    = lane >> 4;
  const int ln   = lane & 15;
  const int mw   = wave >> 1;
  const int nw   = wave & 1;

  v8f zero{};
  v8f acc[2][4];
  for (int mi = 0; mi < 2; ++mi)
    for (int ni = 0; ni < 4; ++ni) acc[mi][ni] = zero;

  const size_t rowA0 = (size_t)blockIdx.y * 128;
  const size_t rowB0 = (size_t)blockIdx.x * 128;
  const unsigned lds_base = (unsigned)(unsigned long long)(void*)smem;
  const int nslab = K / KC;

#if HAVE_TDM
  if (wave == 0) {   // prologue: slab 0 into buffer 0
    tdm_load_2d_bf16(lds_base,               A  + rowA0 * K, KC, 128, K, 4, 7);
    tdm_load_2d_bf16(lds_base + GTB * 2,     Wb + rowB0 * K, KC, 128, K, 4, 7);
  }
#endif

  for (int s = 0; s < nslab; ++s) {
    const int kt = s * KC;
    __syncthreads();            // buf[(s+1)&1] no longer being read
    const bool more = (s + 1 < nslab);
#if HAVE_TDM
    if (wave == 0 && more) {    // async-prefetch next slab into the other buffer
      unsigned off = ((s + 1) & 1) ? (unsigned)(2 * GTB * 2) : 0u;
      tdm_load_2d_bf16(lds_base + off,           A  + rowA0 * K + kt + KC,
                       KC, 128, K, 4, 7);
      tdm_load_2d_bf16(lds_base + off + GTB * 2, Wb + rowB0 * K + kt + KC,
                       KC, 128, K, 4, 7);
    }
    if (wave == 0) {
      if (more) __builtin_amdgcn_s_wait_tensorcnt(2);   // next slab in flight
      else      __builtin_amdgcn_s_wait_tensorcnt(0);
    }
#else
    {
      unsigned short* dA = smem + ((s & 1) ? 2 * GTB : 0);
      unsigned short* dB = dA + GTB;
      for (int i = 0; i < 4; ++i) {
        int c = tid + i * 256;
        int row = c >> 3, col = (c & 7) * 8;
        *(u32x4*)&dA[row * LDK + col] =
            *(const u32x4*)&A[(rowA0 + row) * (size_t)K + kt + col];
        *(u32x4*)&dB[row * LDK + col] =
            *(const u32x4*)&Wb[(rowB0 + row) * (size_t)K + kt + col];
      }
    }
#endif
    __syncthreads();

    const unsigned short* cA = smem + ((s & 1) ? 2 * GTB : 0);
    const unsigned short* cB = cA + GTB;
    for (int kk = 0; kk < KC; kk += 32) {
      v16bf aF[2], bF[4];
      for (int mi = 0; mi < 2; ++mi)
        aF[mi] = load_afrag(&cA[(mw * 32 + mi * 16 + ln) * LDK], kk, g);
      for (int ni = 0; ni < 4; ++ni)
        bF[ni] = load_bfrag(&cB[(nw * 64 + ni * 16 + ln) * LDK], kk, g);
      for (int mi = 0; mi < 2; ++mi)
        for (int ni = 0; ni < 4; ++ni)
          acc[mi][ni] = wmma_bf16(aF[mi], bF[ni], acc[mi][ni]);
    }
  }

  for (int mi = 0; mi < 2; ++mi)
    for (int ni = 0; ni < 4; ++ni)
      for (int r = 0; r < 8; ++r) {
        size_t row = rowA0 + mw * 32 + mi * 16 + g * 8 + r;
        size_t col = rowB0 + nw * 64 + ni * 16 + ln;
        float val = acc[mi][ni][r] + bias[col];
        if (OUT_BF16)
          ((unsigned short*)Cptr)[row * (size_t)N + col] = f2bf(val);
        else
          ((float*)Cptr)[row * (size_t)N + col] = val;
      }
}

// ---------------------------------------------------------------------------
// Kernel 2: 2D-RoPE + scatter qkv[M,12288](bf16) -> q,k,v [B*H][S][128](bf16)
// ---------------------------------------------------------------------------
__global__ __launch_bounds__(256) void rope_scatter(
    const unsigned short* __restrict__ qkv, const int* __restrict__ pos_ids,
    unsigned short* __restrict__ q, unsigned short* __restrict__ k,
    unsigned short* __restrict__ v) {
  int t = blockIdx.x * 256 + threadIdx.x;
  int d = t & 127;
  int h = (t >> 7) & 31;
  int b = (t >> 12) & 1;
  int s = t >> 13;

  size_t base = ((size_t)s * BATCH + b) * QKVN + (size_t)h * 384;
  float qv = bf2f(qkv[base + d]);
  float kv = bf2f(qkv[base + 128 + d]);

  int half = d >> 6;
  int j = d & 63;
  int pos = pos_ids[(size_t)b * 2 * SEQ + (size_t)half * SEQ + s];
  int fi = j & 31;
  float ang = (float)pos * __expf(-(float)fi * 0.2878231366f);  // ln(1e4)/32
  float sn, cs;
  __sincosf(ang, &sn, &cs);

  float rq, rk;
  if (j < 32) {
    rq = -bf2f(qkv[base + d + 32]);
    rk = -bf2f(qkv[base + 128 + d + 32]);
  } else {
    rq = bf2f(qkv[base + d - 32]);
    rk = bf2f(qkv[base + 128 + d - 32]);
  }
  float qo = (qv * cs + rq * sn) * 0.08838834764f;   // fold 1/sqrt(128) into q
  float ko = kv * cs + rk * sn;

  size_t o = (((size_t)b * NHEADS + h) * SEQ + s) * HDIM + d;
  q[o] = f2bf(qo);
  k[o] = f2bf(ko);
  v[o] = qkv[base + 256 + d];
}

// ---------------------------------------------------------------------------
// Kernel 3: flash attention. Q TDM-loaded once; K tiles TDM double-buffered;
// V manually transposed. One live A-frag at a time (stay under 256 VGPRs).
//   LDS: sQ + sK0 + sK1 + sVt + sP, each [128][136] bf16 -> 170KB dynamic
// ---------------------------------------------------------------------------
#define LTS 136                      // LDS tile stride in halves (272B)
#define TILE_BYTES (128 * LTS * 2)

__global__ __launch_bounds__(256) void flash_attn(
    const unsigned short* __restrict__ q, const unsigned short* __restrict__ k,
    const unsigned short* __restrict__ v, unsigned short* __restrict__ ao) {
  extern __shared__ __align__(16) unsigned short smem[];
  unsigned short* sQ  = smem;                  // query tile, row-major
  unsigned short* sK0 = smem + 128 * LTS;      // key tile ping
  unsigned short* sK1 = smem + 2 * 128 * LTS;  // key tile pong
  unsigned short* sV  = smem + 3 * 128 * LTS;  // value tile, transposed [d][t]
  unsigned short* sP  = smem + 4 * 128 * LTS;  // probabilities [row][t]

  const int tid  = threadIdx.x;
  const int wave = tid >> 5;
  const int lane = tid & 31;
  const int g    = lane >> 4;
  const int ln   = lane & 15;
  const int bh   = blockIdx.y;
  const int b    = bh >> 5;
  const int h    = bh & 31;
  const int NT   = SEQ / 128;

  const unsigned lds_base = (unsigned)(unsigned long long)(void*)smem;

#if HAVE_TDM
  if (wave == 0) {
    // Q tile once + first K tile (async; waited inside the loop)
    tdm_load_2d_bf16(lds_base,
                     q + ((size_t)bh * SEQ + (size_t)blockIdx.x * 128) * HDIM,
                     128, 128, 128, 5, 3);
    tdm_load_2d_bf16(lds_base + TILE_BYTES, k + (size_t)bh * SEQ * HDIM,
                     128, 128, 128, 5, 3);
  }
#else
  for (int i = 0; i < 8; ++i) {
    int c = tid + i * 256;
    int row = c >> 4, col = (c & 15) * 8;
    *(u32x4*)&sQ[row * LTS + col] =
        *(const u32x4*)&q[((size_t)bh * SEQ + blockIdx.x * 128 + row) * HDIM + col];
  }
#endif

  v8f zero{};
  v8f of[8];
  for (int dt = 0; dt < 8; ++dt) of[dt] = zero;
  float mrun[8], lrun[8];
  for (int r = 0; r < 8; ++r) { mrun[r] = -1e30f; lrun[r] = 0.f; }

  for (int kt = 0; kt < NT; ++kt) {
    const int t0 = kt * 128;
    __syncthreads();             // prev iteration fully consumed its buffers
    const bool more = (kt + 1 < NT);

#if HAVE_TDM
    if (wave == 0 && more)       // async-prefetch next K tile into other buffer
      tdm_load_2d_bf16(lds_base + (((kt + 1) & 1) ? 2u : 1u) * TILE_BYTES,
                       k + ((size_t)bh * SEQ + t0 + 128) * HDIM,
                       128, 128, 128, 5, 3);
#else
    {
      unsigned short* dK = (kt & 1) ? sK1 : sK0;
      for (int i = 0; i < 8; ++i) {
        int c = tid + i * 256;
        int row = c >> 4, col = (c & 15) * 8;
        *(u32x4*)&dK[row * LTS + col] =
            *(const u32x4*)&k[((size_t)bh * SEQ + t0 + row) * HDIM + col];
      }
    }
#endif
    // ---- stage V tile transposed: sV[d][t] (partial unroll: low VGPR spike)
#pragma unroll 4
    for (int i = 0; i < 16; ++i) {
      int c = tid + i * 256;
      int row = c >> 5, d0 = (c & 31) * 4;
      u16x4 hv = *(const u16x4*)&v[((size_t)bh * SEQ + t0 + row) * HDIM + d0];
      for (int j = 0; j < 4; ++j) sV[(d0 + j) * LTS + row] = hv[j];
    }
#if HAVE_TDM
    if (wave == 0) {
      if (more) __builtin_amdgcn_s_wait_tensorcnt(1);   // next K stays in flight
      else      __builtin_amdgcn_s_wait_tensorcnt(0);
    }
#endif
    __syncthreads();

    const unsigned short* sK = (kt & 1) ? sK1 : sK0;

    // ---- S = Q * K^T : one A-frag live at a time ----
    v8f sf[8];
    for (int nt = 0; nt < 8; ++nt) sf[nt] = zero;
    for (int kk4 = 0; kk4 < 4; ++kk4) {
      v16bf a = load_afrag(&sQ[(wave * 16 + ln) * LTS], kk4 * 32, g);
      for (int nt = 0; nt < 8; ++nt)
        sf[nt] = wmma_bf16(a, load_bfrag(&sK[(nt * 16 + ln) * LTS], kk4 * 32, g),
                           sf[nt]);
    }

    // ---- online softmax (wave32: xor-shuffles within 16-lane halves) ----
    for (int r = 0; r < 8; ++r) {
      float mx = sf[0][r];
      for (int nt = 1; nt < 8; ++nt) mx = fmaxf(mx, sf[nt][r]);
      for (int off = 1; off < 16; off <<= 1) mx = fmaxf(mx, __shfl_xor(mx, off, 32));
      float mnew  = fmaxf(mrun[r], mx);
      float alpha = __expf(mrun[r] - mnew);
      float rs = 0.f;
      for (int nt = 0; nt < 8; ++nt) {
        float p = __expf(sf[nt][r] - mnew);
        sf[nt][r] = p;
        rs += p;
      }
      for (int off = 1; off < 16; off <<= 1) rs += __shfl_xor(rs, off, 32);
      lrun[r] = lrun[r] * alpha + rs;
      mrun[r] = mnew;
      for (int dt = 0; dt < 8; ++dt) of[dt][r] *= alpha;
    }

    // ---- write P (bf16) to per-wave LDS strip (same-wave DS ops in-order) ----
    for (int nt = 0; nt < 8; ++nt)
      for (int r = 0; r < 8; ++r)
        sP[(wave * 16 + g * 8 + r) * LTS + nt * 16 + ln] = f2bf(sf[nt][r]);

    // ---- O += P * V : one A-frag live at a time ----
    for (int kk4 = 0; kk4 < 4; ++kk4) {
      v16bf ap = load_afrag(&sP[(wave * 16 + ln) * LTS], kk4 * 32, g);
      for (int dt = 0; dt < 8; ++dt)
        of[dt] = wmma_bf16(ap, load_bfrag(&sV[(dt * 16 + ln) * LTS], kk4 * 32, g),
                           of[dt]);
    }
  }

  // ---- normalize, store bf16 into [S, B, D] for the output projection ----
  for (int r = 0; r < 8; ++r) {
    float inv = 1.0f / lrun[r];
    int s = blockIdx.x * 128 + wave * 16 + g * 8 + r;
    for (int dt = 0; dt < 8; ++dt) {
      size_t col = (size_t)h * HDIM + dt * 16 + ln;
      ao[((size_t)s * BATCH + b) * DMODEL + col] = f2bf(of[dt][r] * inv);
    }
  }
}

// ---------------------------------------------------------------------------
// Host launcher
// ---------------------------------------------------------------------------
extern "C" void kernel_launch(void* const* d_in, const int* in_sizes, int n_in,
                              void* d_out, int out_size, void* d_ws, size_t ws_size,
                              hipStream_t stream) {
  const float* x    = (const float*)d_in[0];
  const int*   pos  = (const int*)d_in[1];
  const float* wqkv = (const float*)d_in[2];
  const float* bqkv = (const float*)d_in[3];
  const float* wout = (const float*)d_in[4];
  const float* bout = (const float*)d_in[5];
  float* out = (float*)d_out;

  // workspace layout (bf16 halves)
  unsigned short* xb    = (unsigned short*)d_ws;                    // [M, D]
  unsigned short* wqkvb = xb + (size_t)MROWS * DMODEL;              // [3D, D]
  unsigned short* woutb = wqkvb + (size_t)QKVN * DMODEL;            // [D, D]
  unsigned short* qkv   = woutb + (size_t)DMODEL * DMODEL;          // [M, 3D]
  unsigned short* qb    = qkv + (size_t)MROWS * QKVN;               // [B*H][S][128]
  unsigned short* kb    = qb + (size_t)BATCH * NHEADS * SEQ * HDIM;
  unsigned short* vb    = kb + (size_t)BATCH * NHEADS * SEQ * HDIM;
  unsigned short* ao    = vb + (size_t)BATCH * NHEADS * SEQ * HDIM; // [M, D]

  // 0) one-time fp32 -> bf16 down-conversion (x + both weight matrices)
  cvt_f32_bf16<<<(MROWS * (size_t)DMODEL) / 1024, 256, 0, stream>>>(x, xb);
  cvt_f32_bf16<<<((size_t)QKVN * DMODEL) / 1024, 256, 0, stream>>>(wqkv, wqkvb);
  cvt_f32_bf16<<<((size_t)DMODEL * DMODEL) / 1024, 256, 0, stream>>>(wout, woutb);

  size_t gemm_smem = (size_t)4 * GTB * 2;   // 80KB double-buffered tiles

  // 1) QKV projection (TDM-fed, double-buffered bf16 WMMA)
  gemm_wmma<true><<<dim3(QKVN / 128, MROWS / 128), 256, gemm_smem, stream>>>(
      xb, wqkvb, bqkv, qkv, MROWS, QKVN, DMODEL);

  // 2) 2D RoPE + scatter into attention-friendly layout
  rope_scatter<<<(SEQ * BATCH * NHEADS * HDIM) / 256, 256, 0, stream>>>(
      qkv, pos, qb, kb, vb);

  // 3) Flash attention per (b,h), 128 query rows per block
  size_t flash_smem = (size_t)5 * TILE_BYTES;   // 170KB (<= 320KB WGP LDS)
  flash_attn<<<dim3(SEQ / 128, BATCH * NHEADS), 256, flash_smem, stream>>>(
      qb, kb, vb, ao);

  // 4) Output projection -> fp32 out
  gemm_wmma<false><<<dim3(DMODEL / 128, MROWS / 128), 256, gemm_smem, stream>>>(
      ao, woutb, bout, out, MROWS, DMODEL, DMODEL);
}
